// GatedGraphConvModel_69544110457566
// MI455X (gfx1250) — matrix operand order repaired and checked by reading, not compiled, not run on hardware
//
#include <hip/hip_runtime.h>
#include <math.h>

typedef __attribute__((ext_vector_type(16))) __bf16 v16bf;
typedef __attribute__((ext_vector_type(8)))  float  v8f;

#define NNODES 50000
#define NEDGES 800000
#define DIN    256
#define HDIM   128
#define LCLS   6
#define NPAD   50048   // round_up(50000, 128)

// LDS byte-offset of a generic pointer that points into LDS
template <typename T>
__device__ __forceinline__ unsigned lds_off(T* p) {
  return (unsigned)(unsigned long long)(__attribute__((address_space(3))) T*)p;
}

// ---------------- elementwise / utility kernels ----------------

__global__ void k_zero_f32(float* p, long n) {
  long i = (long)blockIdx.x * blockDim.x + threadIdx.x;
  if (i < n) p[i] = 0.f;
}

__global__ void k_cvt_bf16(const float* __restrict__ in, __bf16* __restrict__ out, long n) {
  long i = (long)blockIdx.x * blockDim.x + threadIdx.x;
  if (i < n) out[i] = (__bf16)in[i];
}

// convert [rows_valid x cols] f32 -> [NPADrows x cols] bf16, zero padding rows
__global__ void k_cvt_bf16_pad(const float* __restrict__ in, __bf16* __restrict__ out,
                               int rows_valid, int cols, long ntot) {
  long i = (long)blockIdx.x * blockDim.x + threadIdx.x;
  if (i >= ntot) return;
  long r = i / cols;
  out[i] = (r < rows_valid) ? (__bf16)in[i] : (__bf16)0.f;
}

// in: [R][C] f32 -> out: [C][R] bf16  (weights become [Nout][K] row-major)
__global__ void k_transpose_cvt(const float* __restrict__ in, __bf16* __restrict__ out,
                                int R, int C) {
  long i = (long)blockIdx.x * blockDim.x + threadIdx.x;
  if (i >= (long)R * C) return;
  int c  = (int)(i / R);
  int rr = (int)(i % R);
  out[i] = (__bf16)in[(long)rr * C + c];
}

// ---------------- WMMA BF16 GEMM ----------------
// C[M,Nc] = A[M,K] * Bt[Nc,K]^T (+bias). Block = 256 threads (8 wave32s),
// tile 128(M) x 128(N); K staged in 128-wide LDS chunks via async copies.
// Per wave: 32(M) x 64(N) = 2x4 WMMA tiles -> 8 v_wmma per 32-K step.

template <int K, bool HAS_BIAS, bool HAS_BF16OUT>
__global__ __launch_bounds__(256) void k_gemm_bf16_wmma(
    const __bf16* __restrict__ A,   // [M][K], M = NPAD rows
    const __bf16* __restrict__ Bt,  // [Nc][K]
    float* __restrict__ C,          // [M][Nc]
    __bf16* __restrict__ Cb,        // optional bf16 copy of C
    const float* __restrict__ bias, // optional [Nc]
    int Nc)
{
  constexpr int KC = 128;                       // K chunk held in LDS
  __shared__ __align__(16) __bf16 As[128 * KC]; // 32 KB
  __shared__ __align__(16) __bf16 Bs[128 * KC]; // 32 KB

  const int tid  = threadIdx.x;
  const int lane = tid & 31;
  const int r    = lane & 15;
  const int half = lane >> 4;
  const int wave = tid >> 5;
  const int wm   = wave & 3;    // 4 waves down M: 32 rows each
  const int wn   = wave >> 2;   // 2 waves across N: 64 cols each
  const long m_blk = (long)blockIdx.y * 128;
  const int  n_blk = blockIdx.x * 128;

  const __bf16* gA = A  + m_blk * K;
  const __bf16* gB = Bt + (long)n_blk * K;
  const unsigned lA = lds_off(&As[0]);
  const unsigned lB = lds_off(&Bs[0]);

  v8f acc[2][4] = {};

  #pragma unroll
  for (int kc = 0; kc < K; kc += KC) {
    if (kc != 0) __syncthreads();   // protect LDS reuse across chunks

    // ---- async stage A chunk [128][KC] and B chunk [128][KC] (16B per op) ----
    #pragma unroll
    for (int i = 0; i < (128 * KC / 8) / 256; ++i) {      // 8 iters
      int e   = (tid + i * 256) * 8;                      // element in chunk
      int row = e / KC;
      int col = e % KC;
      const __bf16* ga = gA + (long)row * K + kc + col;
      const __bf16* gb = gB + (long)row * K + kc + col;
      asm volatile("global_load_async_to_lds_b128 %0, %1, off"
                   :: "v"(lA + (unsigned)e * 2), "v"(ga) : "memory");
      asm volatile("global_load_async_to_lds_b128 %0, %1, off"
                   :: "v"(lB + (unsigned)e * 2), "v"(gb) : "memory");
    }
    asm volatile("s_wait_asynccnt 0x0" ::: "memory");
    __syncthreads();

    // ---- compute: 4 x 32-K WMMA steps over the chunk ----
    #pragma unroll
    for (int ks = 0; ks < KC; ks += 32) {
      v16bf af[2];
      v16bf bfr[4];
      #pragma unroll
      for (int tm = 0; tm < 2; ++tm) {
        const __bf16* ap = &As[(wm * 32 + tm * 16 + r) * KC + ks];
        #pragma unroll
        for (int i = 0; i < 8; ++i) {
          af[tm][i]     = ap[half * 8 + i];        // K = half*8 + i
          af[tm][8 + i] = ap[16 + half * 8 + i];   // K = 16 + half*8 + i
        }
      }
      #pragma unroll
      for (int tn = 0; tn < 4; ++tn) {
        const __bf16* bp = &Bs[(wn * 64 + tn * 16 + r) * KC + ks];
        #pragma unroll
        for (int i = 0; i < 8; ++i) {
          bfr[tn][i]     = bp[half * 8 + i];       // K = half*8 + i
          bfr[tn][8 + i] = bp[16 + half * 8 + i];  // K = 16 + half*8 + i
        }
      }
      #pragma unroll
      for (int tm = 0; tm < 2; ++tm)
        #pragma unroll
        for (int tn = 0; tn < 4; ++tn)
          acc[tm][tn] = __builtin_amdgcn_wmma_f32_16x16x32_bf16(
              false, af[tm], false, bfr[tn], (short)0, acc[tm][tn], false, false);
    }
  }

  // ---- epilogue: C/D layout -> VGPR j holds (M = half*8 + j, N = r) ----
  #pragma unroll
  for (int tm = 0; tm < 2; ++tm) {
    #pragma unroll
    for (int tn = 0; tn < 4; ++tn) {
      int col = n_blk + wn * 64 + tn * 16 + r;
      float bv = HAS_BIAS ? bias[col] : 0.f;
      #pragma unroll
      for (int j = 0; j < 8; ++j) {
        long row = m_blk + wm * 32 + tm * 16 + half * 8 + j;
        float v = acc[tm][tn][j] + bv;
        C[row * Nc + col] = v;
        if constexpr (HAS_BF16OUT) Cb[row * Nc + col] = (__bf16)v;
      }
    }
  }
}

// ---------------- edge scatter-add (L2-resident atomics) ----------------
__global__ void k_scatter_add(const float* __restrict__ m, const int* __restrict__ ei,
                              float* __restrict__ agg, long ntot) {
  long i = (long)blockIdx.x * blockDim.x + threadIdx.x;  // E * (HDIM/4)
  if (i >= ntot) return;
  int e = (int)(i >> 5);
  int c = (int)(i & 31) << 2;
  int s = ei[e];
  int d = ei[NEDGES + e];
  const float4 v = *(const float4*)(m + (long)s * HDIM + c);
  float* p = agg + (long)d * HDIM + c;
  atomicAdd(p + 0, v.x);
  atomicAdd(p + 1, v.y);
  atomicAdd(p + 2, v.z);
  atomicAdd(p + 3, v.w);
}

// ---------------- fused GRU gates (+ optional ELU) ----------------
__global__ void k_gru(const float* __restrict__ gi, const float* __restrict__ gh,
                      const float* __restrict__ hin, float* __restrict__ hout,
                      __bf16* __restrict__ hb, int do_elu, long ntot) {
  long i = (long)blockIdx.x * blockDim.x + threadIdx.x;
  if (i >= ntot) return;
  long n = i >> 7;
  int  f = (int)(i & 127);
  long b = n * (3 * HDIM);
  float ir = gi[b + f], iz = gi[b + HDIM + f], inn = gi[b + 2 * HDIM + f];
  float hr = gh[b + f], hz = gh[b + HDIM + f], hnn = gh[b + 2 * HDIM + f];
  float h  = hin[i];
  float rg = 1.f / (1.f + __expf(-(ir + hr)));
  float zg = 1.f / (1.f + __expf(-(iz + hz)));
  float ng = tanhf(inn + rg * hnn);
  float o  = (1.f - zg) * ng + zg * h;
  if (do_elu) o = (o > 0.f) ? o : (__expf(o) - 1.f);
  hout[i] = o;
  if (hb) hb[i] = (__bf16)o;
}

// ---------------- classifier (K=128 -> L=6, tiny) ----------------
__global__ void k_classifier(const float* __restrict__ h, const float* __restrict__ w,
                             const float* __restrict__ b, float* __restrict__ logits,
                             long ntot) {
  long i = (long)blockIdx.x * blockDim.x + threadIdx.x;
  if (i >= ntot) return;
  int n = (int)(i / LCLS);
  int c = (int)(i % LCLS);
  const float* hr = h + (long)n * HDIM;
  float s = b[c];
  #pragma unroll 8
  for (int k = 0; k < HDIM; ++k) s += hr[k] * w[k * LCLS + c];
  logits[i] = s;
}

__global__ void k_zero_loss(float* loss) { *loss = 0.f; }

__global__ void k_loss(const float* __restrict__ logits, const int* __restrict__ y,
                       float* __restrict__ loss, int n_nodes) {
  int n = blockIdx.x * blockDim.x + threadIdx.x;
  if (n >= n_nodes) return;
  const float* l = logits + (long)n * LCLS;
  float mx = l[0];
  #pragma unroll
  for (int c = 1; c < LCLS; ++c) mx = fmaxf(mx, l[c]);
  float s = 0.f;
  #pragma unroll
  for (int c = 0; c < LCLS; ++c) s += __expf(l[c] - mx);
  float lp = l[y[n]] - mx - __logf(s);
  atomicAdd(loss, -lp / (float)n_nodes);
}

__global__ void k_copy_f32(const float* __restrict__ in, float* __restrict__ out, long n) {
  long i = (long)blockIdx.x * blockDim.x + threadIdx.x;
  if (i < n) out[i] = in[i];
}

// ---------------- orchestration ----------------

static inline long cdiv(long a, long b) { return (a + b - 1) / b; }

extern "C" void kernel_launch(void* const* d_in, const int* in_sizes, int n_in,
                              void* d_out, int out_size, void* d_ws, size_t ws_size,
                              hipStream_t stream) {
  (void)in_sizes; (void)n_in; (void)out_size; (void)ws_size;

  const float* x     = (const float*)d_in[0];
  const int*   ei    = (const int*)  d_in[1];
  const int*   y     = (const int*)  d_in[2];
  const float* lin_w = (const float*)d_in[3];
  const float* lin_b = (const float*)d_in[4];
  const float* w1    = (const float*)d_in[5];
  const float* wih1  = (const float*)d_in[6];
  const float* whh1  = (const float*)d_in[7];
  const float* bih1  = (const float*)d_in[8];
  const float* bhh1  = (const float*)d_in[9];
  const float* w2    = (const float*)d_in[10];
  const float* wih2  = (const float*)d_in[11];
  const float* whh2  = (const float*)d_in[12];
  const float* bih2  = (const float*)d_in[13];
  const float* bhh2  = (const float*)d_in[14];
  const float* cls_w = (const float*)d_in[15];
  const float* cls_b = (const float*)d_in[16];

  // workspace carve-up
  char* p = (char*)d_ws;
  auto alloc = [&](size_t bytes) -> void* {
    void* q = (void*)p;
    p += (bytes + 255) & ~(size_t)255;
    return q;
  };
  // all B matrices stored [Nout][K] bf16
  __bf16* wb_linT = (__bf16*)alloc((size_t)HDIM * DIN * 2);       // [128][256]
  __bf16* wb_w1T  = (__bf16*)alloc((size_t)HDIM * HDIM * 2);      // [128][128]
  __bf16* wb_w2T  = (__bf16*)alloc((size_t)HDIM * HDIM * 2);
  __bf16* wb_ih1  = (__bf16*)alloc((size_t)3 * HDIM * HDIM * 2);  // [384][128] (torch layout)
  __bf16* wb_hh1  = (__bf16*)alloc((size_t)3 * HDIM * HDIM * 2);
  __bf16* wb_ih2  = (__bf16*)alloc((size_t)3 * HDIM * HDIM * 2);
  __bf16* wb_hh2  = (__bf16*)alloc((size_t)3 * HDIM * HDIM * 2);
  __bf16* xb      = (__bf16*)alloc((size_t)NPAD * DIN * 2);
  __bf16* hb      = (__bf16*)alloc((size_t)NPAD * HDIM * 2);
  __bf16* aggb    = (__bf16*)alloc((size_t)NPAD * HDIM * 2);
  float*  h_f     = (float*) alloc((size_t)NPAD * HDIM * 4);
  float*  m_f     = (float*) alloc((size_t)NPAD * HDIM * 4);
  float*  agg_f   = (float*) alloc((size_t)NPAD * HDIM * 4);
  float*  gi_f    = (float*) alloc((size_t)NPAD * 3 * HDIM * 4);
  float*  gh_f    = (float*) alloc((size_t)NPAD * 3 * HDIM * 4);

  float* out_logits   = (float*)d_out;                      // [N, 6]
  float* out_loss     = out_logits + (long)NNODES * LCLS;   // [1]
  float* out_features = out_loss + 1;                       // [N, 128]

  const int T = 256;

  // ---- weight prep ----
  k_transpose_cvt<<<cdiv((long)DIN * HDIM, T), T, 0, stream>>>(lin_w, wb_linT, DIN, HDIM);
  k_transpose_cvt<<<cdiv((long)HDIM * HDIM, T), T, 0, stream>>>(w1, wb_w1T, HDIM, HDIM);
  k_transpose_cvt<<<cdiv((long)HDIM * HDIM, T), T, 0, stream>>>(w2, wb_w2T, HDIM, HDIM);
  k_cvt_bf16<<<cdiv((long)3 * HDIM * HDIM, T), T, 0, stream>>>(wih1, wb_ih1, (long)3 * HDIM * HDIM);
  k_cvt_bf16<<<cdiv((long)3 * HDIM * HDIM, T), T, 0, stream>>>(whh1, wb_hh1, (long)3 * HDIM * HDIM);
  k_cvt_bf16<<<cdiv((long)3 * HDIM * HDIM, T), T, 0, stream>>>(wih2, wb_ih2, (long)3 * HDIM * HDIM);
  k_cvt_bf16<<<cdiv((long)3 * HDIM * HDIM, T), T, 0, stream>>>(whh2, wb_hh2, (long)3 * HDIM * HDIM);

  // ---- x -> bf16 padded ----
  k_cvt_bf16_pad<<<cdiv((long)NPAD * DIN, T), T, 0, stream>>>(x, xb, NNODES, DIN, (long)NPAD * DIN);

  // ---- input projection: h = x @ lin_w + lin_b ----
  k_gemm_bf16_wmma<DIN, true, true>
      <<<dim3(HDIM / 128, NPAD / 128), T, 0, stream>>>(xb, wb_linT, h_f, hb, lin_b, HDIM);

  const __bf16* wmsgT[2] = {wb_w1T, wb_w2T};
  const __bf16* wih[2]   = {wb_ih1, wb_ih2};
  const __bf16* whh[2]   = {wb_hh1, wb_hh2};
  const float*  bih[2]   = {bih1, bih2};
  const float*  bhh[2]   = {bhh1, bhh2};

  for (int layer = 0; layer < 2; ++layer) {
    // m = h @ w
    k_gemm_bf16_wmma<HDIM, false, false>
        <<<dim3(HDIM / 128, NPAD / 128), T, 0, stream>>>(hb, wmsgT[layer], m_f,
                                                         (__bf16*)0, (const float*)0, HDIM);
    // agg = scatter_add(m[src] -> dst)
    k_zero_f32<<<cdiv((long)NPAD * HDIM, T), T, 0, stream>>>(agg_f, (long)NPAD * HDIM);
    k_scatter_add<<<cdiv((long)NEDGES * (HDIM / 4), T), T, 0, stream>>>(
        m_f, ei, agg_f, (long)NEDGES * (HDIM / 4));
    k_cvt_bf16<<<cdiv((long)NPAD * HDIM, T), T, 0, stream>>>(agg_f, aggb, (long)NPAD * HDIM);
    // gi = agg @ wih.T + bih ; gh = h @ whh.T + bhh
    k_gemm_bf16_wmma<HDIM, true, false>
        <<<dim3(3 * HDIM / 128, NPAD / 128), T, 0, stream>>>(aggb, wih[layer], gi_f,
                                                             (__bf16*)0, bih[layer], 3 * HDIM);
    k_gemm_bf16_wmma<HDIM, true, false>
        <<<dim3(3 * HDIM / 128, NPAD / 128), T, 0, stream>>>(hb, whh[layer], gh_f,
                                                             (__bf16*)0, bhh[layer], 3 * HDIM);
    // GRU combine (+ ELU after layer 0); refresh bf16 h for next GEMMs
    k_gru<<<cdiv((long)NPAD * HDIM, T), T, 0, stream>>>(gi_f, gh_f, h_f, h_f,
                                                        (layer == 0) ? hb : (__bf16*)0,
                                                        (layer == 0) ? 1 : 0,
                                                        (long)NPAD * HDIM);
  }

  // ---- outputs ----
  k_classifier<<<cdiv((long)NNODES * LCLS, T), T, 0, stream>>>(h_f, cls_w, cls_b, out_logits,
                                                               (long)NNODES * LCLS);
  k_copy_f32<<<cdiv((long)NNODES * HDIM, T), T, 0, stream>>>(h_f, out_features,
                                                             (long)NNODES * HDIM);
  k_zero_loss<<<1, 1, 0, stream>>>(out_loss);
  k_loss<<<cdiv(NNODES, T), T, 0, stream>>>(out_logits, y, out_loss, NNODES);
}
